// MetricalLinkPredictionModel_8083128451633
// MI455X (gfx1250) — matrix operand order repaired and checked by reading, not compiled, not run on hardware
//
#include <hip/hip_runtime.h>
#include <hip/hip_bf16.h>

// ---------------- problem constants (from reference) ----------------
#define N_NODES 100000
#define N_REL   7
#define E_REL   800000
#define E_TGT   500000
#define IN_F    128
#define HID     256
#define ECHUNK  125000   // 4 chunks of target edges
#define NCHUNK  4

// ---------------- WMMA vector types (gfx1250, wave32) ----------------
typedef __attribute__((ext_vector_type(16))) __bf16 v16bf;
typedef __attribute__((ext_vector_type(8)))  __bf16 v8bf;
typedef __attribute__((ext_vector_type(8)))  float  v8f;

// GEMM tiling: 128x128 C tile per 256-thread (8-wave) workgroup, BK=32.
// Each wave owns a 32x64 subtile = 2x4 fragments of 16x16.
// LDS tiles are double-buffered (2 x 20KB) to overlap global loads with WMMA.
#define BM 128
#define BN 128
#define BK 32
#define LDT 40   // padded LDS row length in bf16 (80B, 16B aligned, bank-spread)

__device__ __forceinline__ v8bf pack8(float4 a, float4 b) {
  v8bf p;
  p[0]=(__bf16)a.x; p[1]=(__bf16)a.y; p[2]=(__bf16)a.z; p[3]=(__bf16)a.w;
  p[4]=(__bf16)b.x; p[5]=(__bf16)b.y; p[6]=(__bf16)b.z; p[7]=(__bf16)b.w;
  return p;
}

// Fragment assembly per CDNA5 ISA layouts:
//  A (16x32 bf16): lane l<16 -> row l, K {0..7,16..23}; lane>=16 -> row l-16, K {8..15,24..31}
//  B (32x16 bf16): lane l<16 -> col l, K {0..7,16..23}; lane>=16 -> col l-16, K {8..15,24..31}
// All 6 fragments are preloaded into distinct registers so the 8 WMMAs are
// mutually independent and can issue back-to-back (no full DScnt drains).
__device__ __forceinline__ void wmma_frags(const __bf16* sA, const __bf16* sB,
                                           int wm, int wn, int l, v8f acc[2][4]) {
  const int r  = l & 15;
  const int ak = (l < 16) ? 0 : 8;
  const int bk = (l < 16) ? 0 : 8;
  v16bf af[2], bfv[4];
#pragma unroll
  for (int mf = 0; mf < 2; ++mf) {
    const __bf16* p = sA + (wm + mf*16 + r) * LDT + ak;
    v8bf lo = *(const v8bf*)(p);
    v8bf hi = *(const v8bf*)(p + 16);
    v16bf f;
#pragma unroll
    for (int i = 0; i < 8; ++i) { f[i] = lo[i]; f[i+8] = hi[i]; }
    af[mf] = f;
  }
#pragma unroll
  for (int nf = 0; nf < 4; ++nf) {
    const __bf16* p = sB + (wn + nf*16 + r) * LDT + bk;
    v8bf lo = *(const v8bf*)p;
    v8bf hi = *(const v8bf*)(p + 16);
    v16bf f;
#pragma unroll
    for (int i = 0; i < 8; ++i) { f[i] = lo[i]; f[i+8] = hi[i]; }
    bfv[nf] = f;
  }
#pragma unroll
  for (int nf = 0; nf < 4; ++nf) {
    acc[0][nf] = __builtin_amdgcn_wmma_f32_16x16x32_bf16(false, af[0], false, bfv[nf],
                                                         (short)0, acc[0][nf], false, false);
    acc[1][nf] = __builtin_amdgcn_wmma_f32_16x16x32_bf16(false, af[1], false, bfv[nf],
                                                         (short)0, acc[1][nf], false, false);
  }
}

// C[M,N] += A_f32[M,K](lda) @ B, with B supplied transposed as Bt[N][K] bf16.
// C is pre-initialized (kernel loads C into accumulators first) -> sequential
// launches on the same C accumulate (used for the per-relation sum).
__global__ __launch_bounds__(256) void gemm_acc(const float* __restrict__ A, int lda,
                                                const __bf16* __restrict__ Bt,
                                                float* __restrict__ C,
                                                int M, int N, int K) {
  __shared__ __bf16 sA[2][BM * LDT];
  __shared__ __bf16 sB[2][BN * LDT];
  const int tid = threadIdx.x;
  const int l = tid & 31, w = tid >> 5;
  const int wm = (w & 3) * 32, wn = (w >> 2) * 64;
  const int m0 = blockIdx.y * BM, n0 = blockIdx.x * BN;
  const int llo = l & 15, mofs = (l < 16) ? 0 : 8;

  v8f acc[2][4];
#pragma unroll
  for (int mf = 0; mf < 2; ++mf)
#pragma unroll
    for (int nf = 0; nf < 4; ++nf) {
      const int gn  = n0 + wn + nf*16 + llo;
      const int gmb = m0 + wm + mf*16 + mofs;
      v8f a;
#pragma unroll
      for (int j = 0; j < 8; ++j)
        a[j] = (gmb + j < M) ? C[(size_t)(gmb + j) * N + gn] : 0.f;
      acc[mf][nf] = a;
    }

  const int arow  = tid >> 1;          // 0..127 (tile row for A and Bt)
  const int apart = (tid & 1) * 16;    // 16-element K chunk
  int gm = m0 + arow; if (gm >= M) gm = M - 1;   // clamped; OOB rows never stored
  const float*  aptr = A  + (size_t)gm * lda + apart;
  const __bf16* bptr = Bt + (size_t)(n0 + arow) * K + apart;
  const int sOff = arow * LDT + apart;

  // prologue: stage tile 0
  {
    const float4* a4 = (const float4*)aptr;
    float4 f0 = a4[0], f1 = a4[1], f2 = a4[2], f3 = a4[3];
    *(v8bf*)&sA[0][sOff]     = pack8(f0, f1);
    *(v8bf*)&sA[0][sOff + 8] = pack8(f2, f3);
    const v8bf* b8 = (const v8bf*)bptr;
    *(v8bf*)&sB[0][sOff]     = b8[0];
    *(v8bf*)&sB[0][sOff + 8] = b8[1];
  }
  __syncthreads();

  int cur = 0;
  for (int k0 = 0; k0 < K; k0 += BK) {
    const int nxt = cur ^ 1;
    const bool more = (k0 + BK) < K;
    v8bf na0 = {}, na1 = {}, nb0 = {}, nb1 = {};
    if (more) {
      const float4* a4 = (const float4*)(aptr + k0 + BK);
      float4 f0 = a4[0], f1 = a4[1], f2 = a4[2], f3 = a4[3];
      na0 = pack8(f0, f1); na1 = pack8(f2, f3);
      const v8bf* b8 = (const v8bf*)(bptr + k0 + BK);
      nb0 = b8[0]; nb1 = b8[1];
      if (k0 + 2*BK < K) __builtin_prefetch(aptr + k0 + 2*BK, 0, 3); // global_prefetch_b8
    }
    wmma_frags(sA[cur], sB[cur], wm, wn, l, acc);   // overlaps in-flight global loads
    if (more) {
      *(v8bf*)&sA[nxt][sOff]     = na0;
      *(v8bf*)&sA[nxt][sOff + 8] = na1;
      *(v8bf*)&sB[nxt][sOff]     = nb0;
      *(v8bf*)&sB[nxt][sOff + 8] = nb1;
    }
    __syncthreads();
    cur = nxt;
  }

#pragma unroll
  for (int mf = 0; mf < 2; ++mf)
#pragma unroll
    for (int nf = 0; nf < 4; ++nf) {
      const int gn  = n0 + wn + nf*16 + llo;
      const int gmb = m0 + wm + mf*16 + mofs;
#pragma unroll
      for (int j = 0; j < 8; ++j)
        if (gmb + j < M) C[(size_t)(gmb + j) * N + gn] = acc[mf][nf][j];
    }
}

// Head GEMM: A row e = [ H[src[e]] | H[dst[e]] ] (K = 2*HID), gathered on the fly.
__global__ __launch_bounds__(256) void gemm_gather_acc(const float* __restrict__ H,
                                                       const long long* __restrict__ gs,
                                                       const long long* __restrict__ gd,
                                                       const __bf16* __restrict__ Bt,
                                                       float* __restrict__ C,
                                                       int M, int N, int K) {
  __shared__ __bf16 sA[2][BM * LDT];
  __shared__ __bf16 sB[2][BN * LDT];
  const int tid = threadIdx.x;
  const int l = tid & 31, w = tid >> 5;
  const int wm = (w & 3) * 32, wn = (w >> 2) * 64;
  const int m0 = blockIdx.y * BM, n0 = blockIdx.x * BN;
  const int llo = l & 15, mofs = (l < 16) ? 0 : 8;

  v8f acc[2][4];
#pragma unroll
  for (int mf = 0; mf < 2; ++mf)
#pragma unroll
    for (int nf = 0; nf < 4; ++nf) {
      const int gn  = n0 + wn + nf*16 + llo;
      const int gmb = m0 + wm + mf*16 + mofs;
      v8f a;
#pragma unroll
      for (int j = 0; j < 8; ++j)
        a[j] = (gmb + j < M) ? C[(size_t)(gmb + j) * N + gn] : 0.f;
      acc[mf][nf] = a;
    }

  const int arow  = tid >> 1;
  const int apart = (tid & 1) * 16;
  int e = m0 + arow; if (e >= M) e = 0;
  const long long s = gs[e], d = gd[e];
  const float* rowS = H + (size_t)s * HID;
  const float* rowD = H + (size_t)d * HID;
  const __bf16* bptr = Bt + (size_t)(n0 + arow) * K + apart;
  const int sOff = arow * LDT + apart;

  // prologue: stage tile 0 (col 0..15 always from the src half)
  {
    const float4* a4 = (const float4*)(rowS + apart);
    float4 f0 = a4[0], f1 = a4[1], f2 = a4[2], f3 = a4[3];
    *(v8bf*)&sA[0][sOff]     = pack8(f0, f1);
    *(v8bf*)&sA[0][sOff + 8] = pack8(f2, f3);
    const v8bf* b8 = (const v8bf*)bptr;
    *(v8bf*)&sB[0][sOff]     = b8[0];
    *(v8bf*)&sB[0][sOff + 8] = b8[1];
  }
  __syncthreads();

  int cur = 0;
  for (int k0 = 0; k0 < K; k0 += BK) {
    const int nxt = cur ^ 1;
    const bool more = (k0 + BK) < K;
    v8bf na0 = {}, na1 = {}, nb0 = {}, nb1 = {};
    if (more) {
      const int col = k0 + BK + apart;               // 16-wide chunk never straddles HID
      const float* src = (col < HID) ? (rowS + col) : (rowD + (col - HID));
      const float4* a4 = (const float4*)src;
      float4 f0 = a4[0], f1 = a4[1], f2 = a4[2], f3 = a4[3];
      na0 = pack8(f0, f1); na1 = pack8(f2, f3);
      const v8bf* b8 = (const v8bf*)(bptr + k0 + BK);
      nb0 = b8[0]; nb1 = b8[1];
    }
    wmma_frags(sA[cur], sB[cur], wm, wn, l, acc);
    if (more) {
      *(v8bf*)&sA[nxt][sOff]     = na0;
      *(v8bf*)&sA[nxt][sOff + 8] = na1;
      *(v8bf*)&sB[nxt][sOff]     = nb0;
      *(v8bf*)&sB[nxt][sOff + 8] = nb1;
    }
    __syncthreads();
    cur = nxt;
  }

#pragma unroll
  for (int mf = 0; mf < 2; ++mf)
#pragma unroll
    for (int nf = 0; nf < 4; ++nf) {
      const int gn  = n0 + wn + nf*16 + llo;
      const int gmb = m0 + wm + mf*16 + mofs;
#pragma unroll
      for (int j = 0; j < 8; ++j)
        if (gmb + j < M) C[(size_t)(gmb + j) * N + gn] = acc[mf][nf][j];
    }
}

// ---------------- scatter / utility kernels ----------------
__global__ void zero_f32(float* __restrict__ p, long long n4) {
  long long t = (long long)blockIdx.x * blockDim.x + threadIdx.x;
  if (t < n4) ((float4*)p)[t] = make_float4(0.f, 0.f, 0.f, 0.f);
}

// segment-sum: AGG[dst] += X[src], one relation at a time (f32 L2 atomics)
__global__ void scatter_add(const long long* __restrict__ ei, int rel,
                            const float* __restrict__ X, float* __restrict__ AGG, int C) {
  const int per = C >> 2;
  long long t = (long long)blockIdx.x * blockDim.x + threadIdx.x;
  if (t >= (long long)E_REL * per) return;
  const int e = (int)(t / per);
  const int j = (int)(t % per) * 4;
  const long long* base = ei + (size_t)rel * 2 * E_REL;
  const int s = (int)base[e];
  const int d = (int)base[E_REL + e];
  float4 v = *(const float4*)(X + (size_t)s * C + j);
  float* dst = AGG + (size_t)d * C + j;
  atomicAdd(dst + 0, v.x);
  atomicAdd(dst + 1, v.y);
  atomicAdd(dst + 2, v.z);
  atomicAdd(dst + 3, v.w);
}

// dst[r][n*K+k] = bf16(src[r][k*N+n])   (weight transpose+convert)
__global__ void tcvt(const float* __restrict__ src, __bf16* __restrict__ dst,
                     int K, int N, int R) {
  long long t = (long long)blockIdx.x * blockDim.x + threadIdx.x;
  const long long mat = (long long)K * N;
  if (t >= (long long)R * mat) return;
  const int r = (int)(t / mat);
  const long long rem = t % mat;
  const int n = (int)(rem / K), k = (int)(rem % K);
  dst[(size_t)r * mat + (size_t)n * K + k] = (__bf16)src[(size_t)r * mat + (size_t)k * N + n];
}

// dst[n*K+k] = bf16( sum_r src[r][k*N+n] )   (collapse Wr over relations)
__global__ void tcvt_sum(const float* __restrict__ src, __bf16* __restrict__ dst,
                         int K, int N, int R) {
  long long t = (long long)blockIdx.x * blockDim.x + threadIdx.x;
  const long long mat = (long long)K * N;
  if (t >= mat) return;
  const int n = (int)(t / K), k = (int)(t % K);
  float s = 0.f;
  for (int r = 0; r < R; ++r) s += src[(size_t)r * mat + (size_t)k * N + n];
  dst[(size_t)n * K + k] = (__bf16)s;
}

__global__ void bias_sum(const float* __restrict__ b, float* __restrict__ o, int C, int R) {
  const int j = threadIdx.x + blockIdx.x * blockDim.x;
  if (j >= C) return;
  float s = 0.f;
  for (int r = 0; r < R; ++r) s += b[(size_t)r * C + j];
  o[j] = s;
}

// h = relu(l2norm((acc + bsum)/7)), one wave per 256-wide row, in place
__global__ __launch_bounds__(256) void epi_norm_relu(float* __restrict__ H,
                                                     const float* __restrict__ bs, int rows) {
  const int row = blockIdx.x * 8 + (threadIdx.x >> 5);
  const int l = threadIdx.x & 31;
  if (row >= rows) return;
  float* p = H + (size_t)row * HID + l * 8;
  const float* bp = bs + l * 8;
  float t[8]; float ss = 0.f;
#pragma unroll
  for (int i = 0; i < 8; ++i) { t[i] = (p[i] + bp[i]) * (1.f / 7.f); ss += t[i] * t[i]; }
  for (int o = 16; o; o >>= 1) ss += __shfl_xor(ss, o, 32);
  const float inv = 1.f / fmaxf(sqrtf(ss), 1e-12f);
#pragma unroll
  for (int i = 0; i < 8; ++i) p[i] = fmaxf(t[i] * inv, 0.f);
}

// h2 = (acc + bsum)/7 (raw last layer)
__global__ void epi_mean(float* __restrict__ H, const float* __restrict__ bs, long long total) {
  long long t = (long long)blockIdx.x * blockDim.x + threadIdx.x;
  if (t >= total) return;
  H[t] = (H[t] + bs[t % HID]) * (1.f / 7.f);
}

// z1 = relu(z1 + pitch*w[512] + onset0*w[513] + onset1*w[514] + pb1)
__global__ void epi_head1(float* __restrict__ Z, const float* __restrict__ pitch,
                          const float* __restrict__ onset, const float* __restrict__ pw1,
                          const float* __restrict__ pb1, int eoff, int ecnt) {
  long long t = (long long)blockIdx.x * blockDim.x + threadIdx.x;
  if (t >= (long long)ecnt * HID) return;
  const int e = (int)(t / HID), j = (int)(t % HID);
  const int ge = eoff + e;
  float v = Z[t] + pitch[ge]       * pw1[512 * HID + j]
                 + onset[2 * ge]   * pw1[513 * HID + j]
                 + onset[2 * ge+1] * pw1[514 * HID + j]
                 + pb1[j];
  Z[t] = fmaxf(v, 0.f);
}

__global__ void epi_relu_bias(float* __restrict__ Z, const float* __restrict__ b,
                              int C, long long total) {
  long long t = (long long)blockIdx.x * blockDim.x + threadIdx.x;
  if (t >= total) return;
  Z[t] = fmaxf(Z[t] + b[t % C], 0.f);
}

// out = sigmoid(z2 . pw3 + pb3), one wave per edge
__global__ __launch_bounds__(256) void head_final(const float* __restrict__ Z2,
                                                  const float* __restrict__ pw3,
                                                  const float* __restrict__ pb3,
                                                  float* __restrict__ out, int eoff, int ecnt) {
  const int e = blockIdx.x * 8 + (threadIdx.x >> 5);
  const int l = threadIdx.x & 31;
  if (e >= ecnt) return;
  const float4 z = *(const float4*)(Z2 + (size_t)e * 128 + l * 4);
  const float4 w = *(const float4*)(pw3 + l * 4);
  float s = z.x * w.x + z.y * w.y + z.z * w.z + z.w * w.w;
  for (int o = 16; o; o >>= 1) s += __shfl_xor(s, o, 32);
  if (l == 0) out[eoff + e] = 1.f / (1.f + expf(-(s + pb3[0])));
}

// ---------------- host orchestration ----------------
extern "C" void kernel_launch(void* const* d_in, const int* in_sizes, int n_in,
                              void* d_out, int out_size, void* d_ws, size_t ws_size,
                              hipStream_t stream) {
  const float*     xv    = (const float*)d_in[0];
  const long long* ei    = (const long long*)d_in[1];
  const float*     Wl1   = (const float*)d_in[2];
  const float*     Wr1   = (const float*)d_in[3];
  const float*     b1v   = (const float*)d_in[4];
  const float*     Wl2   = (const float*)d_in[5];
  const float*     Wr2   = (const float*)d_in[6];
  const float*     b2v   = (const float*)d_in[7];
  const long long* tei   = (const long long*)d_in[8];
  const float*     pitchv= (const float*)d_in[9];
  const float*     onsetv= (const float*)d_in[10];
  const float*     pw1v  = (const float*)d_in[11];
  const float*     pb1v  = (const float*)d_in[12];
  const float*     pw2v  = (const float*)d_in[13];
  const float*     pb2v  = (const float*)d_in[14];
  const float*     pw3v  = (const float*)d_in[15];
  const float*     pb3v  = (const float*)d_in[16];
  float* out = (float*)d_out;
  (void)in_sizes; (void)n_in; (void)out_size; (void)ws_size;

  char* wsb = (char*)d_ws;
  size_t cur = 0;
  auto alloc = [&](size_t bytes) -> void* {
    void* p = wsb + cur;
    cur += (bytes + 255) & ~(size_t)255;
    return p;
  };
  float* agg = (float*)alloc((size_t)N_NODES * HID * 4);      // per-relation segment sum
  float* h1  = (float*)alloc((size_t)N_NODES * HID * 4);      // layer1 acc -> normalized h1
  float* h2  = (float*)alloc((size_t)N_NODES * HID * 4);      // layer2 acc -> raw h2
  float* z1  = (float*)alloc((size_t)ECHUNK * HID * 4);       // head stage 1 (per chunk)
  float* z2  = (float*)alloc((size_t)ECHUNK * (HID/2) * 4);   // head stage 2 (per chunk)
  __bf16* Bt1wr = (__bf16*)alloc((size_t)HID * IN_F * 2);
  __bf16* Bt1wl = (__bf16*)alloc((size_t)N_REL * HID * IN_F * 2);
  __bf16* Bt2wr = (__bf16*)alloc((size_t)HID * HID * 2);
  __bf16* Bt2wl = (__bf16*)alloc((size_t)N_REL * HID * HID * 2);
  __bf16* BtH1  = (__bf16*)alloc((size_t)HID * (2*HID) * 2);
  __bf16* BtH2  = (__bf16*)alloc((size_t)(HID/2) * HID * 2);
  float* bsum1 = (float*)alloc(HID * 4);
  float* bsum2 = (float*)alloc(HID * 4);

  const int TB = 256;
  auto nb = [](long long n, int tb) { return (unsigned)((n + tb - 1) / tb); };

  // --- weight prep: transpose to Bt[N][K] bf16; collapse Wr and biases over relations ---
  tcvt_sum<<<nb((long long)IN_F * HID, TB), TB, 0, stream>>>(Wr1, Bt1wr, IN_F, HID, N_REL);
  tcvt    <<<nb((long long)N_REL * IN_F * HID, TB), TB, 0, stream>>>(Wl1, Bt1wl, IN_F, HID, N_REL);
  tcvt_sum<<<nb((long long)HID * HID, TB), TB, 0, stream>>>(Wr2, Bt2wr, HID, HID, N_REL);
  tcvt    <<<nb((long long)N_REL * HID * HID, TB), TB, 0, stream>>>(Wl2, Bt2wl, HID, HID, N_REL);
  tcvt    <<<nb((long long)(2*HID) * HID, TB), TB, 0, stream>>>(pw1v, BtH1, 2*HID, HID, 1);
  tcvt    <<<nb((long long)HID * (HID/2), TB), TB, 0, stream>>>(pw2v, BtH2, HID, HID/2, 1);
  bias_sum<<<1, HID, 0, stream>>>(b1v, bsum1, HID, N_REL);
  bias_sum<<<1, HID, 0, stream>>>(b2v, bsum2, HID, N_REL);

  const dim3 gNode(HID / BN, (N_NODES + BM - 1) / BM);

  // --- layer 1: h1 = relu(l2norm((x@Wr1sum + sum_r agg_r@Wl1_r + bsum1)/7)) ---
  zero_f32<<<nb((long long)N_NODES * HID / 4, TB), TB, 0, stream>>>(h1, (long long)N_NODES * HID / 4);
  gemm_acc<<<gNode, 256, 0, stream>>>(xv, IN_F, Bt1wr, h1, N_NODES, HID, IN_F);
  for (int r = 0; r < N_REL; ++r) {
    zero_f32<<<nb((long long)N_NODES * IN_F / 4, TB), TB, 0, stream>>>(agg, (long long)N_NODES * IN_F / 4);
    scatter_add<<<nb((long long)E_REL * (IN_F / 4), TB), TB, 0, stream>>>(ei, r, xv, agg, IN_F);
    gemm_acc<<<gNode, 256, 0, stream>>>(agg, IN_F, Bt1wl + (size_t)r * HID * IN_F, h1, N_NODES, HID, IN_F);
  }
  epi_norm_relu<<<(N_NODES + 7) / 8, 256, 0, stream>>>(h1, bsum1, N_NODES);

  // --- layer 2: h2 = (h1@Wr2sum + sum_r agg_r@Wl2_r + bsum2)/7 (raw) ---
  zero_f32<<<nb((long long)N_NODES * HID / 4, TB), TB, 0, stream>>>(h2, (long long)N_NODES * HID / 4);
  gemm_acc<<<gNode, 256, 0, stream>>>(h1, HID, Bt2wr, h2, N_NODES, HID, HID);
  for (int r = 0; r < N_REL; ++r) {
    zero_f32<<<nb((long long)N_NODES * HID / 4, TB), TB, 0, stream>>>(agg, (long long)N_NODES * HID / 4);
    scatter_add<<<nb((long long)E_REL * (HID / 4), TB), TB, 0, stream>>>(ei, r, h1, agg, HID);
    gemm_acc<<<gNode, 256, 0, stream>>>(agg, HID, Bt2wl + (size_t)r * HID * HID, h2, N_NODES, HID, HID);
  }
  epi_mean<<<nb((long long)N_NODES * HID, TB), TB, 0, stream>>>(h2, bsum2, (long long)N_NODES * HID);

  // --- link-prediction head, in 4 edge chunks ---
  const long long* tsrc = tei;
  const long long* tdst = tei + E_TGT;
  for (int c = 0; c < NCHUNK; ++c) {
    const int eoff = c * ECHUNK;
    const int ecnt = ECHUNK;
    const dim3 gh1(HID / BN, (ecnt + BM - 1) / BM);
    const dim3 gh2((HID/2) / BN, (ecnt + BM - 1) / BM);
    zero_f32<<<nb((long long)ecnt * HID / 4, TB), TB, 0, stream>>>(z1, (long long)ecnt * HID / 4);
    gemm_gather_acc<<<gh1, 256, 0, stream>>>(h2, tsrc + eoff, tdst + eoff, BtH1, z1, ecnt, HID, 2*HID);
    epi_head1<<<nb((long long)ecnt * HID, TB), TB, 0, stream>>>(z1, pitchv, onsetv, pw1v, pb1v, eoff, ecnt);
    zero_f32<<<nb((long long)ecnt * (HID/2) / 4, TB), TB, 0, stream>>>(z2, (long long)ecnt * (HID/2) / 4);
    gemm_acc<<<gh2, 256, 0, stream>>>(z1, HID, BtH2, z2, ecnt, HID/2, HID);
    epi_relu_bias<<<nb((long long)ecnt * (HID/2), TB), TB, 0, stream>>>(z2, pb2v, HID/2, (long long)ecnt * (HID/2));
    head_final<<<(ecnt + 7) / 8, 256, 0, stream>>>(z2, pw3v, pb3v, out, eoff, ecnt);
  }
}
